// Model_82532091560585
// MI455X (gfx1250) — compile-verified, matrix operand order checked by reading, not run on hardware
//
#include <hip/hip_runtime.h>

#define ND 10000
#define NP 20000
#define DH 128
#define NE 1000000
#define NL 500000

typedef __attribute__((ext_vector_type(2)))  float  v2f;
typedef __attribute__((ext_vector_type(8)))  float  v8f;
typedef __attribute__((ext_vector_type(16))) __bf16 v16bf;

// ---------------------------------------------------------------- zero
__global__ __launch_bounds__(256) void zero4_kernel(float4* __restrict__ p, int n4) {
  int i = blockIdx.x * 256 + threadIdx.x;
  if (i < n4) p[i] = make_float4(0.f, 0.f, 0.f, 0.f);
}

// ------------------------------------------------- edge scatter (both dirs)
// agg_b[col] += xa[row];  agg_a[row] += xb[col]   (per 128-feature row)
// Batches of 32 edges: coalesced index load + shuffle broadcast.
__global__ __launch_bounds__(256) void edge_scatter_kernel(
    const float* __restrict__ xa, const float* __restrict__ xb,
    const int* __restrict__ rows, const int* __restrict__ cols,
    float* __restrict__ agg_b, float* __restrict__ agg_a, int n_edges) {
  int lane  = threadIdx.x & 31;
  int wave  = (blockIdx.x * blockDim.x + threadIdx.x) >> 5;
  int nwav  = (gridDim.x * blockDim.x) >> 5;
  int f     = lane * 4;  // 32 lanes x float4 = 128 features
  for (int e0 = wave * 32; e0 < n_edges; e0 += nwav * 32) {
    int nb  = min(32, n_edges - e0);
    int r_l = (lane < nb) ? rows[e0 + lane] : 0;
    int c_l = (lane < nb) ? cols[e0 + lane] : 0;
    for (int j = 0; j < nb; j++) {
      int r = __shfl(r_l, j, 32);
      int c = __shfl(c_l, j, 32);
      float4 va = *(const float4*)(xa + (size_t)r * DH + f);
      float4 vb = *(const float4*)(xb + (size_t)c * DH + f);
      float* pb = agg_b + (size_t)c * DH + f;
      atomicAdd(pb + 0, va.x); atomicAdd(pb + 1, va.y);
      atomicAdd(pb + 2, va.z); atomicAdd(pb + 3, va.w);
      float* pa = agg_a + (size_t)r * DH + f;
      atomicAdd(pa + 0, vb.x); atomicAdd(pa + 1, vb.y);
      atomicAdd(pa + 2, vb.z); atomicAdd(pa + 3, vb.w);
    }
  }
}

// ---------------- pre-pack a [128,128] f32 weight into WMMA-B v2f fragments
// frag layout: [wave(8)][kstep(32)][lane(32)][pair(2)]  (16384 floats)
__global__ __launch_bounds__(256) void prep_wfrag_f32_kernel(
    const float* __restrict__ W, float* __restrict__ frag) {
  int i = blockIdx.x * 256 + threadIdx.x;  // 16384 total
  int p = i & 1, lane = (i >> 1) & 31, ks = (i >> 6) & 31, wv = i >> 11;
  int lr = lane & 15, hi = lane >> 4;
  int K = ks * 4 + 2 * hi + p;
  int col = wv * 16 + lr;
  frag[i] = W[K * DH + col];
}

// ------------------------------- SAGE dense part: out = act(agg@Wl + x@Wr + b)
// One block = 16 rows x 128 cols; 8 waves, each owns a 16x16 tile.
// FP32 WMMA 16x16x4. A from padded LDS, B from pre-packed fragments (1 b64/WMMA).
__global__ __launch_bounds__(256) void sage_wmma_kernel(
    const float* __restrict__ agg, const float* __restrict__ x,
    const v2f* __restrict__ WlF, const v2f* __restrict__ WrF,
    const float* __restrict__ bias, float* __restrict__ out,
    int nrows, int do_relu) {
  __shared__ float As[16 * 132];
  __shared__ float Xs[16 * 132];
  const int tid  = threadIdx.x;
  const int row0 = blockIdx.x * 16;

  // cooperative stage of 16x128 A tiles (float4, 512 vec4 per matrix)
  for (int i4 = tid; i4 < 512; i4 += 256) {
    int r = i4 >> 5, c4 = i4 & 31;
    ((float4*)(As + r * 132))[c4] = ((const float4*)(agg + (size_t)row0 * DH))[i4];
    ((float4*)(Xs + r * 132))[c4] = ((const float4*)(x   + (size_t)row0 * DH))[i4];
  }
  __syncthreads();

  const int wave = tid >> 5, lane = tid & 31;
  const int lr = lane & 15, hi = lane >> 4;
  const int col0 = wave * 16;

  v8f c = {0.f, 0.f, 0.f, 0.f, 0.f, 0.f, 0.f, 0.f};
  const float* a_row = As + lr * 132;
  const float* x_row = Xs + lr * 132;
  const v2f* BL = WlF + wave * 1024 + lane;  // + ks*32
  const v2f* BR = WrF + wave * 1024 + lane;
  for (int ks = 0; ks < 32; ks++) {
    v2f a;
    a.x = a_row[4 * ks + 2 * hi + 0];
    a.y = a_row[4 * ks + 2 * hi + 1];
    v2f b = BL[ks * 32];
    c = __builtin_amdgcn_wmma_f32_16x16x4_f32(false, a, false, b, (short)0, c, false, false);
  }
  for (int ks = 0; ks < 32; ks++) {
    v2f a;
    a.x = x_row[4 * ks + 2 * hi + 0];
    a.y = x_row[4 * ks + 2 * hi + 1];
    v2f b = BR[ks * 32];
    c = __builtin_amdgcn_wmma_f32_16x16x4_f32(false, a, false, b, (short)0, c, false, false);
  }

  const float bv = bias[col0 + lr];
#pragma unroll
  for (int v = 0; v < 8; v++) {
    int m = v + 8 * hi;
    float val = c[v] + bv;
    if (do_relu) val = fmaxf(val, 0.f);
    out[(size_t)(row0 + m) * DH + col0 + lr] = val;
  }
}

// ---------------------- pre-pack Wd1 [256,128] f32 -> bf16 WMMA-B fragments
// frag layout: [wave(8)][kstep(8)][lane(32)][elem(16)]
__global__ __launch_bounds__(256) void prep_wd1_kernel(
    const float* __restrict__ Wd1, __bf16* __restrict__ frag) {
  int i = blockIdx.x * 256 + threadIdx.x;  // 32768 total
  int e = i & 15, lane = (i >> 4) & 31, ks = (i >> 9) & 7, wv = i >> 12;
  int lr = lane & 15, hi = lane >> 4;
  int K = ks * 32 + ((e < 8) ? (e + 8 * hi) : (e + 8 + 8 * hi));
  int col = wv * 16 + lr;
  frag[i] = (__bf16)Wd1[K * DH + col];
}

// -------------------------------------------------- decoder (BF16 WMMA)
// 16 labels per block: gather z -> bf16 LDS, H = relu([z_d|z_p] @ Wd1 + bd1),
// out = H @ Wd2 + bd2 via half-wave shuffle reduction.
#define ZP 264
__global__ __launch_bounds__(256) void decoder_wmma_kernel(
    const float* __restrict__ z_drug, const float* __restrict__ z_prot,
    const int* __restrict__ lrow, const int* __restrict__ lcol,
    const __bf16* __restrict__ wfrag, const float* __restrict__ bd1,
    const float* __restrict__ Wd2, const float* __restrict__ bd2,
    float* __restrict__ outp, int nlab) {
  __shared__ __bf16 Zs[16 * ZP];
  __shared__ float red[16];
  const int tid = threadIdx.x;
  const int l0  = blockIdx.x * 16;

  {  // gather + f32->bf16 convert: thread t -> label t>>4, 16-feature chunk t&15
    int lab = tid >> 4, ch = tid & 15;
    int li = l0 + lab;
    if (li < nlab) {
      const float* src = (ch < 8)
          ? (z_drug + (size_t)lrow[li] * DH + ch * 16)
          : (z_prot + (size_t)lcol[li] * DH + (ch - 8) * 16);
      __bf16* dst = Zs + lab * ZP + ch * 16;
#pragma unroll
      for (int j = 0; j < 16; j += 4) {
        float4 v = *(const float4*)(src + j);
        dst[j + 0] = (__bf16)v.x; dst[j + 1] = (__bf16)v.y;
        dst[j + 2] = (__bf16)v.z; dst[j + 3] = (__bf16)v.w;
      }
    }
  }
  if (tid < 16) red[tid] = 0.f;
  __syncthreads();

  const int wave = tid >> 5, lane = tid & 31;
  const int lr = lane & 15, hi = lane >> 4;
  const int col0 = wave * 16;
  const v16bf* bptr = (const v16bf*)wfrag;

  v8f c = {0.f, 0.f, 0.f, 0.f, 0.f, 0.f, 0.f, 0.f};
  const __bf16* zr = Zs + lr * ZP;
  for (int ks = 0; ks < 8; ks++) {
    v16bf a;
#pragma unroll
    for (int j = 0; j < 8; j++) {  // element pairs (2j,2j+1) are K-consecutive
      int ka = ks * 32 + ((j < 4) ? (2 * j + 8 * hi) : (2 * j + 8 + 8 * hi));
      a[2 * j]     = zr[ka];
      a[2 * j + 1] = zr[ka + 1];
    }
    v16bf b = bptr[(wave * 8 + ks) * 32 + lane];
    c = __builtin_amdgcn_wmma_f32_16x16x32_bf16(false, a, false, b, (short)0, c, false, false);
  }

  const int col = col0 + lr;
  const float bb = bd1[col];
  const float w2 = Wd2[col];
#pragma unroll
  for (int v = 0; v < 8; v++) {
    int m = v + 8 * hi;                       // label row within tile
    float h = fmaxf(c[v] + bb, 0.f);
    float p = h * w2;                         // partial over this wave's 16 cols
    p += __shfl_xor(p, 1, 32);
    p += __shfl_xor(p, 2, 32);
    p += __shfl_xor(p, 4, 32);
    p += __shfl_xor(p, 8, 32);                // reduced within each 16-lane half
    if (lr == 0) atomicAdd(&red[m], p);
  }
  __syncthreads();
  if (tid < 16) {
    int li = l0 + tid;
    if (li < nlab) outp[li] = red[tid] + bd2[0];
  }
}

// ---------------------------------------------------------------- launch
extern "C" void kernel_launch(void* const* d_in, const int* in_sizes, int n_in,
                              void* d_out, int out_size, void* d_ws, size_t ws_size,
                              hipStream_t stream) {
  const float* x_drug = (const float*)d_in[0];
  const float* x_prot = (const float*)d_in[1];
  const int*   ei_row = (const int*)d_in[2];
  const int*   ei_col = (const int*)d_in[3];
  const int*   el_row = (const int*)d_in[4];
  const int*   el_col = (const int*)d_in[5];
  const float* W1f_l = (const float*)d_in[6];
  const float* b1f   = (const float*)d_in[7];
  const float* W1f_r = (const float*)d_in[8];
  const float* W1r_l = (const float*)d_in[9];
  const float* b1r   = (const float*)d_in[10];
  const float* W1r_r = (const float*)d_in[11];
  const float* W2f_l = (const float*)d_in[12];
  const float* b2f   = (const float*)d_in[13];
  const float* W2f_r = (const float*)d_in[14];
  const float* W2r_l = (const float*)d_in[15];
  const float* b2r   = (const float*)d_in[16];
  const float* W2r_r = (const float*)d_in[17];
  const float* Wd1   = (const float*)d_in[18];
  const float* bd1   = (const float*)d_in[19];
  const float* Wd2   = (const float*)d_in[20];
  const float* bd2   = (const float*)d_in[21];

  // workspace layout
  float*  agg_prot = (float*)d_ws;                    // NP*128
  float*  agg_drug = agg_prot + (size_t)NP * DH;      // ND*128
  float*  h_prot   = agg_drug + (size_t)ND * DH;      // NP*128
  float*  h_drug   = h_prot   + (size_t)NP * DH;      // ND*128
  __bf16* wfragD   = (__bf16*)(h_drug + (size_t)ND * DH);  // 32768 bf16
  float*  wf       = (float*)(wfragD + 32768);        // 8 x 16384 f32 fragments
  float*  F1f_l = wf + 0 * 16384;
  float*  F1f_r = wf + 1 * 16384;
  float*  F1r_l = wf + 2 * 16384;
  float*  F1r_r = wf + 3 * 16384;
  float*  F2f_l = wf + 4 * 16384;
  float*  F2f_r = wf + 5 * 16384;
  float*  F2r_l = wf + 6 * 16384;
  float*  F2r_r = wf + 7 * 16384;

  // output layout: (z_drug, z_prot, out)
  float* z_drug = (float*)d_out;
  float* z_prot = z_drug + (size_t)ND * DH;
  float* out_s  = z_prot + (size_t)NP * DH;

  const int nAgg4 = (NP + ND) * DH / 4;  // agg_prot+agg_drug are contiguous

  // weight fragment packs (independent of everything else)
  prep_wd1_kernel<<<128, 256, 0, stream>>>(Wd1, wfragD);
  prep_wfrag_f32_kernel<<<64, 256, 0, stream>>>(W1f_l, F1f_l);
  prep_wfrag_f32_kernel<<<64, 256, 0, stream>>>(W1f_r, F1f_r);
  prep_wfrag_f32_kernel<<<64, 256, 0, stream>>>(W1r_l, F1r_l);
  prep_wfrag_f32_kernel<<<64, 256, 0, stream>>>(W1r_r, F1r_r);
  prep_wfrag_f32_kernel<<<64, 256, 0, stream>>>(W2f_l, F2f_l);
  prep_wfrag_f32_kernel<<<64, 256, 0, stream>>>(W2f_r, F2f_r);
  prep_wfrag_f32_kernel<<<64, 256, 0, stream>>>(W2r_l, F2r_l);
  prep_wfrag_f32_kernel<<<64, 256, 0, stream>>>(W2r_r, F2r_r);

  // Layer 1
  zero4_kernel<<<(nAgg4 + 255) / 256, 256, 0, stream>>>((float4*)agg_prot, nAgg4);
  edge_scatter_kernel<<<2048, 256, 0, stream>>>(x_drug, x_prot, ei_row, ei_col,
                                                agg_prot, agg_drug, NE);
  sage_wmma_kernel<<<NP / 16, 256, 0, stream>>>(agg_prot, x_prot, (const v2f*)F1f_l,
                                                (const v2f*)F1f_r, b1f, h_prot, NP, 1);
  sage_wmma_kernel<<<ND / 16, 256, 0, stream>>>(agg_drug, x_drug, (const v2f*)F1r_l,
                                                (const v2f*)F1r_r, b1r, h_drug, ND, 1);

  // Layer 2 (reuse agg buffers)
  zero4_kernel<<<(nAgg4 + 255) / 256, 256, 0, stream>>>((float4*)agg_prot, nAgg4);
  edge_scatter_kernel<<<2048, 256, 0, stream>>>(h_drug, h_prot, ei_row, ei_col,
                                                agg_prot, agg_drug, NE);
  sage_wmma_kernel<<<NP / 16, 256, 0, stream>>>(agg_prot, h_prot, (const v2f*)F2f_l,
                                                (const v2f*)F2f_r, b2f, z_prot, NP, 0);
  sage_wmma_kernel<<<ND / 16, 256, 0, stream>>>(agg_drug, h_drug, (const v2f*)F2r_l,
                                                (const v2f*)F2r_r, b2r, z_drug, ND, 0);

  // Decoder
  decoder_wmma_kernel<<<NL / 16, 256, 0, stream>>>(z_drug, z_prot, el_row, el_col,
                                                   wfragD, bd1, Wd2, bd2, out_s, NL);
}
